// SlotAttention_35768487641284
// MI455X (gfx1250) — compile-verified, matrix operand order checked by reading, not run on hardware
//
#include <hip/hip_runtime.h>

// ---------------------------------------------------------------------------
// Slot Attention for MI455X (gfx1250): f16 WMMA GEMMs + fp32 LN/softmax/GRU.
// v is stored transposed (v_t[d][t]) so both big attention GEMM streams are
// contiguous along K; k/v tiles leave the LDS via async-from-LDS stores.
// ---------------------------------------------------------------------------

typedef _Float16 half_t;
typedef __attribute__((ext_vector_type(16))) _Float16 v16h;
typedef __attribute__((ext_vector_type(8)))  float    v8f;

#define DINL static __device__ __forceinline__

// Set to 0 if the assembler rejects the async-from-LDS mnemonic.
#define SA_USE_ASYNC_LDS_STORE 1

constexpr int B_ = 64;
constexpr int T_ = 4096;
constexpr int D_ = 256;
constexpr int S_ = 8;

DINL v8f zero8() {
  v8f z;
#pragma unroll
  for (int i = 0; i < 8; ++i) z[i] = 0.0f;
  return z;
}

DINL v16h zero16h() {
  v16h z;
#pragma unroll
  for (int i = 0; i < 16; ++i) z[i] = (half_t)0.0f;
  return z;
}

DINL v8f wmma16(v16h a, v16h b, v8f c) {
  // D = A(16x32 f16) * B(32x16 f16) + C(16x16 f32)
  return __builtin_amdgcn_wmma_f32_16x16x32_f16(false, a, false, b, (short)0, c,
                                                false, false);
}

// Fragment from row-major f16 [16 rows x >=32 cols], row stride ldm (halves).
// ISA 7.12.2: lane L -> row L&15, K-base (L&16)?8:0; halves 0-7 = K+0..7,
// halves 8-15 = K+16..23.
DINL v16h frag_h(const half_t* base, int ldm, int lane) {
  const int m  = lane & 15;
  const int kb = (lane & 16) ? 8 : 0;
  const half_t* p = base + (size_t)m * ldm + kb;
  v16h a;
#pragma unroll
  for (int i = 0; i < 8; ++i) a[i] = p[i];
#pragma unroll
  for (int i = 0; i < 8; ++i) a[8 + i] = p[16 + i];
  return a;
}

// A fragment built from fp32 LDS (attention weights); rows >= 8 are zero.
DINL v16h frag_a_f32(const float* base, int ldm, int lane) {
  const int m  = lane & 15;
  const int kb = (lane & 16) ? 8 : 0;
  v16h a = zero16h();
  if (m < 8) {
    const float* p = base + (size_t)m * ldm + kb;
#pragma unroll
    for (int i = 0; i < 8; ++i) a[i] = (half_t)p[i];
#pragma unroll
    for (int i = 0; i < 8; ++i) a[8 + i] = (half_t)p[16 + i];
  }
  return a;
}

DINL float wred_sum(float v) {
#pragma unroll
  for (int off = 16; off > 0; off >>= 1) v += __shfl_xor(v, off, 32);
  return v;
}
DINL float wred_max(float v) {
#pragma unroll
  for (int off = 16; off > 0; off >>= 1) v = fmaxf(v, __shfl_xor(v, off, 32));
  return v;
}

#if SA_USE_ASYNC_LDS_STORE
// 128-bit async LDS -> global store (tracked by ASYNCcnt).
DINL void async_store_b128(unsigned long long gaddr, unsigned int lds_off) {
  asm volatile("global_store_async_from_lds_b128 %0, %1, off"
               :: "v"(gaddr), "v"(lds_off) : "memory");
}
DINL void wait_asynccnt0() {
  asm volatile("s_wait_asynccnt 0x0" ::: "memory");
}
#endif

// ---------------------------------------------------------------------------
// Kernel: fp32 -> f16 conversion (weights)
// ---------------------------------------------------------------------------
__global__ void sa_cvt_f16(const float* __restrict__ src,
                           half_t* __restrict__ dst, int n) {
  int i = blockIdx.x * 256 + threadIdx.x;
  if (i < n) dst[i] = (half_t)src[i];
}

// ---------------------------------------------------------------------------
// Kernel: slots = mu + exp(logsigma) * noise
// ---------------------------------------------------------------------------
__global__ void sa_init_slots(const float* __restrict__ noise,
                              const float* __restrict__ mu,
                              const float* __restrict__ ls,
                              float* __restrict__ slots) {
  int i = blockIdx.x * 256 + threadIdx.x;  // over 64*8*256 = 131072
  if (i < B_ * S_ * D_) {
    int sd = i & (S_ * D_ - 1);
    slots[i] = mu[sd] + __expf(ls[sd]) * noise[i];
  }
}

// ---------------------------------------------------------------------------
// Kernel: x = LN(inputs); k = x@Wk^T (row-major [t][e]);
//         v_t = (x@Wv^T)^T (d-major [d][t]).  f16 outputs.
// grid: (32 t-tiles, 64 batches), block 256 (8 waves).
// LDS: xh 128x264h | kt 128x256h | vt 256x136h  (~198 KB)
// ---------------------------------------------------------------------------
constexpr int    LDX      = 264;  // padded x row stride (halves)
constexpr int    VT_LD    = 136;  // padded vt row stride (halves)
constexpr size_t OFF_XH   = 0;                       // 67584 B
constexpr size_t OFF_KT   = 67584;                   // 65536 B
constexpr size_t OFF_VT   = 133120;                  // 69632 B
constexpr size_t SMEM_KV  = 202752;

__global__ void sa_ln_kv(const float* __restrict__ inputs,
                         const float* __restrict__ gamma,
                         const float* __restrict__ beta,
                         const half_t* __restrict__ Wk_h,
                         const half_t* __restrict__ Wv_h,
                         half_t* __restrict__ k_h,
                         half_t* __restrict__ v_t) {
  extern __shared__ char smem[];
  half_t* xh = (half_t*)(smem + OFF_XH);  // [128][LDX]
  half_t* kt = (half_t*)(smem + OFF_KT);  // [128][256]
  half_t* vt = (half_t*)(smem + OFF_VT);  // [256][VT_LD]

  const int n    = blockIdx.y;
  const int tile = blockIdx.x;            // 128-row tile of t
  const int tid  = threadIdx.x;
  const int lane = tid & 31;
  const int wv   = tid >> 5;

  const size_t inBase = ((size_t)n * T_ + (size_t)tile * 128) * D_;

  // ---- LayerNorm 128 rows, wave per row (interleaved) ----
  for (int r = wv; r < 128; r += 8) {
    const float* row = inputs + inBase + (size_t)r * D_;
    float x[8];
    float s = 0.f, s2 = 0.f;
#pragma unroll
    for (int i = 0; i < 8; ++i) {
      x[i] = row[lane * 8 + i];
      s += x[i];
      s2 += x[i] * x[i];
    }
    s  = wred_sum(s);
    s2 = wred_sum(s2);
    const float mu  = s * (1.0f / D_);
    const float var = s2 * (1.0f / D_) - mu * mu;
    const float rs  = rsqrtf(var + 1e-5f);
    half_t* dst = xh + (size_t)r * LDX + lane * 8;
#pragma unroll
    for (int i = 0; i < 8; ++i) {
      const int c = lane * 8 + i;
      dst[i] = (half_t)((x[i] - mu) * rs * gamma[c] + beta[c]);
    }
  }
  __syncthreads();

  // ---- GEMM: each wave owns 16 rows; 16 N-tiles; K = 256 (8 chunks) ----
  const int r0 = wv * 16;
  for (int nt = 0; nt < 16; ++nt) {
    v8f acck = zero8();
    v8f accv = zero8();
#pragma unroll
    for (int kc = 0; kc < 8; ++kc) {
      v16h a  = frag_h(xh + (size_t)r0 * LDX + kc * 32, LDX, lane);
      v16h bk = frag_h(Wk_h + (size_t)nt * 16 * D_ + kc * 32, D_, lane);
      v16h bv = frag_h(Wv_h + (size_t)nt * 16 * D_ + kc * 32, D_, lane);
      acck = wmma16(a, bk, acck);
      accv = wmma16(a, bv, accv);
    }
    const int col = nt * 16 + (lane & 15);
    const int mb  = (lane < 16) ? 0 : 8;
#pragma unroll
    for (int r = 0; r < 8; ++r) {
      const int m = r0 + mb + r;                   // tile-local t
      kt[(size_t)m * D_ + col]    = (half_t)acck[r];
      vt[(size_t)col * VT_LD + m] = (half_t)accv[r];
    }
  }
  __syncthreads();

  // ---- bulk write-out: LDS -> global, 16B granules ----
  half_t* kDst = k_h + inBase;                     // [128][256] row-major
#if SA_USE_ASYNC_LDS_STORE
  for (int u = tid; u < (128 * D_) / 8; u += 256)  // 4096 x 16B chunks
    async_store_b128((unsigned long long)(kDst + (size_t)u * 8),
                     (unsigned int)(OFF_KT + (size_t)u * 16));
  {
    const int d = tid;                             // one d-row per thread
    half_t* dst = v_t + ((size_t)n * D_ + d) * T_ + (size_t)tile * 128;
    const unsigned int src = (unsigned int)(OFF_VT + (size_t)d * VT_LD * 2);
#pragma unroll
    for (int u = 0; u < 16; ++u)
      async_store_b128((unsigned long long)(dst + u * 8),
                       src + (unsigned int)(u * 16));
  }
  wait_asynccnt0();
#else
  for (int u = tid; u < (128 * D_) / 8; u += 256)
    *(uint4*)(kDst + (size_t)u * 8) = *(const uint4*)(kt + (size_t)u * 8);
  {
    const int d = tid;
    half_t* dst = v_t + ((size_t)n * D_ + d) * T_ + (size_t)tile * 128;
    const half_t* src = vt + (size_t)d * VT_LD;
#pragma unroll
    for (int u = 0; u < 16; ++u)
      *(uint4*)(dst + u * 8) = *(const uint4*)(src + u * 8);
  }
#endif
}

// ---------------------------------------------------------------------------
// Kernel: one slot-attention iteration. grid 64 (batch), block 256 (8 waves).
// ---------------------------------------------------------------------------
constexpr size_t OFF_SL     = 0;                 // float [8][256]    8 KB
constexpr size_t OFF_SLH    = 8192;              // half  [16][256]   8 KB
constexpr size_t OFF_QIN    = 16384;             // half  [16][256]   8 KB
constexpr size_t OFF_QH     = 24576;             // half  [16][256]   8 KB
constexpr size_t OFF_UPDH   = 32768;             // half  [16][256]   8 KB
constexpr size_t OFF_HIDH   = 40960;             // half  [16][256]   8 KB
constexpr size_t OFF_GI     = 49152;             // float [8][768]   24 KB
constexpr size_t OFF_GH     = 73728;             // float [8][768]   24 KB
constexpr size_t OFF_LOGITS = 98304;             // float [8][4096] 128 KB
constexpr size_t SMEM_ITER  = 98304 + 131072;    // 224 KB (< 320 KB WGP LDS)

__global__ void sa_iter(float* __restrict__ slots_g,
                        const half_t* __restrict__ k_h,
                        const half_t* __restrict__ v_t,
                        const half_t* __restrict__ Wq_h,
                        const half_t* __restrict__ Wih_h,
                        const half_t* __restrict__ Whh_h,
                        const half_t* __restrict__ W1_h,
                        const half_t* __restrict__ W2_h,
                        const float* __restrict__ b_ih,
                        const float* __restrict__ b_hh,
                        const float* __restrict__ mlp_b1,
                        const float* __restrict__ mlp_b2,
                        const float* __restrict__ ln_s_g,
                        const float* __restrict__ ln_s_b,
                        const float* __restrict__ ln_m_g,
                        const float* __restrict__ ln_m_b,
                        float* __restrict__ out, int write_out) {
  extern __shared__ char smem[];
  float*  sl     = (float*)(smem + OFF_SL);
  half_t* slh    = (half_t*)(smem + OFF_SLH);
  half_t* qin_h  = (half_t*)(smem + OFF_QIN);
  half_t* qh     = (half_t*)(smem + OFF_QH);
  half_t* updh   = (half_t*)(smem + OFF_UPDH);
  half_t* hidh   = (half_t*)(smem + OFF_HIDH);
  float*  gi     = (float*)(smem + OFF_GI);
  float*  gh     = (float*)(smem + OFF_GH);
  float*  logits = (float*)(smem + OFF_LOGITS);

  const int n    = blockIdx.x;
  const int tid  = threadIdx.x;
  const int lane = tid & 31;
  const int wv   = tid >> 5;

  // ---- A: load slots, f16 copy, zero-pad rows 8..15, LN(slots) -> qin_h ----
  for (int i = tid; i < S_ * D_; i += 256) {
    const float vv = slots_g[(size_t)n * S_ * D_ + i];
    sl[i]  = vv;
    slh[i] = (half_t)vv;
    slh[S_ * D_ + i] = (half_t)0.0f;   // rows 8..15
    qin_h[S_ * D_ + i] = (half_t)0.0f; // rows 8..15
  }
  __syncthreads();
  {
    const int s = wv;  // wave per slot
    const float* row = sl + (size_t)s * D_;
    float x[8];
    float sm = 0.f, sq = 0.f;
#pragma unroll
    for (int i = 0; i < 8; ++i) {
      x[i] = row[lane * 8 + i];
      sm += x[i];
      sq += x[i] * x[i];
    }
    sm = wred_sum(sm);
    sq = wred_sum(sq);
    const float mu  = sm * (1.0f / D_);
    const float var = sq * (1.0f / D_) - mu * mu;
    const float rs  = rsqrtf(var + 1e-5f);
#pragma unroll
    for (int i = 0; i < 8; ++i) {
      const int c = lane * 8 + i;
      qin_h[(size_t)s * D_ + c] = (half_t)((x[i] - mu) * rs * ln_s_g[c] + ln_s_b[c]);
    }
  }
  __syncthreads();

  // ---- B: q = LN(slots) @ Wq^T -> qh (f16, 16 rows, top 8 valid) ----
  for (int nt = wv * 2; nt < wv * 2 + 2; ++nt) {
    v8f acc = zero8();
#pragma unroll
    for (int kc = 0; kc < 8; ++kc) {
      v16h a = frag_h(qin_h + kc * 32, D_, lane);
      v16h b = frag_h(Wq_h + (size_t)nt * 16 * D_ + kc * 32, D_, lane);
      acc = wmma16(a, b, acc);
    }
    const int col = nt * 16 + (lane & 15);
    const int mb  = (lane < 16) ? 0 : 8;
#pragma unroll
    for (int r = 0; r < 8; ++r) qh[(size_t)(mb + r) * D_ + col] = (half_t)acc[r];
  }
  __syncthreads();

  // ---- C: logits = q @ k^T * scale -> LDS fp32 [8][4096] ----
  const half_t* kbase = k_h + (size_t)n * T_ * D_;
  const float scale = 0.0625f;  // 1/sqrt(256)
  for (int tt = wv; tt < 256; tt += 8) {
    const half_t* kp = kbase + (size_t)tt * 16 * D_;
    if (tt + 8 < 256)
      __builtin_prefetch(kbase + (size_t)(tt + 8) * 16 * D_, 0, 1);
    v8f acc = zero8();
#pragma unroll
    for (int kc = 0; kc < 8; ++kc) {
      v16h a = frag_h(qh + kc * 32, D_, lane);
      v16h b = frag_h(kp + kc * 32, D_, lane);  // B[K=d][N=t] = k[t][d]
      acc = wmma16(a, b, acc);
    }
    if (lane < 16) {  // rows 0..7
      const int t = tt * 16 + lane;
#pragma unroll
      for (int r = 0; r < 8; ++r) logits[(size_t)r * T_ + t] = acc[r] * scale;
    }
  }
  __syncthreads();

  // ---- D: softmax over t, wave per slot; attn kept fp32 in-place ----
  {
    float* lrow = logits + (size_t)wv * T_;
    float mx = -1e30f;
    for (int i = lane; i < T_; i += 32) mx = fmaxf(mx, lrow[i]);
    mx = wred_max(mx);
    float sum = 0.f;
    for (int i = lane; i < T_; i += 32) {
      const float e = __expf(lrow[i] - mx);
      lrow[i] = e;
      sum += e;
    }
    sum = wred_sum(sum);
    const float f = (1.0f / sum) * (1.0f / (1.0f + 1e-8f));
    for (int i = lane; i < T_; i += 32) lrow[i] *= f;
  }
  __syncthreads();

  // ---- E: updates = attn @ v -> updh (f16); v_t is [d][t], K contiguous ----
  const half_t* vtbase = v_t + (size_t)n * D_ * T_;
  for (int nt = wv * 2; nt < wv * 2 + 2; ++nt) {
    const half_t* vp = vtbase + (size_t)nt * 16 * T_;
    v8f acc = zero8();
    for (int kt = 0; kt < 128; ++kt) {  // K = 4096 in chunks of 32
      v16h a = frag_a_f32(logits + kt * 32, T_, lane);
      v16h b = frag_h(vp + kt * 32, T_, lane);  // B[K=t][N=d] = v_t[d][t]
      acc = wmma16(a, b, acc);
    }
    const int col = nt * 16 + (lane & 15);
    const int mb  = (lane < 16) ? 0 : 8;
#pragma unroll
    for (int r = 0; r < 8; ++r) updh[(size_t)(mb + r) * D_ + col] = (half_t)acc[r];
  }
  __syncthreads();

  // ---- F: gi = updates@W_ih^T, gh = slots_prev@W_hh^T  (N = 768) ----
  for (int nt = wv * 6; nt < wv * 6 + 6; ++nt) {
    v8f ai = zero8();
    v8f ah = zero8();
#pragma unroll
    for (int kc = 0; kc < 8; ++kc) {
      v16h au = frag_h(updh + kc * 32, D_, lane);
      v16h as = frag_h(slh + kc * 32, D_, lane);
      v16h bi = frag_h(Wih_h + (size_t)nt * 16 * D_ + kc * 32, D_, lane);
      v16h bh = frag_h(Whh_h + (size_t)nt * 16 * D_ + kc * 32, D_, lane);
      ai = wmma16(au, bi, ai);
      ah = wmma16(as, bh, ah);
    }
    if (lane < 16) {
      const int col = nt * 16 + lane;
#pragma unroll
      for (int r = 0; r < 8; ++r) {
        gi[(size_t)r * 768 + col] = ai[r];
        gh[(size_t)r * 768 + col] = ah[r];
      }
    }
  }
  __syncthreads();

  // ---- G: GRU gates (torch order r,z,n) ----
  for (int i = tid; i < S_ * D_; i += 256) {
    const int s = i >> 8, d = i & 255;
    const float gir = gi[(size_t)s * 768 + d] + b_ih[d];
    const float ghr = gh[(size_t)s * 768 + d] + b_hh[d];
    const float giz = gi[(size_t)s * 768 + 256 + d] + b_ih[256 + d];
    const float ghz = gh[(size_t)s * 768 + 256 + d] + b_hh[256 + d];
    const float gin = gi[(size_t)s * 768 + 512 + d] + b_ih[512 + d];
    const float ghn = gh[(size_t)s * 768 + 512 + d] + b_hh[512 + d];
    const float r = 1.0f / (1.0f + __expf(-(gir + ghr)));
    const float z = 1.0f / (1.0f + __expf(-(giz + ghz)));
    const float h = tanhf(gin + r * ghn);
    sl[i] = (1.0f - z) * h + z * sl[i];
  }
  __syncthreads();

  // ---- H: LN(slots) with mlp params -> qin_h (rows 8..15 still zero) ----
  {
    const int s = wv;
    const float* row = sl + (size_t)s * D_;
    float x[8];
    float sm = 0.f, sq = 0.f;
#pragma unroll
    for (int i = 0; i < 8; ++i) {
      x[i] = row[lane * 8 + i];
      sm += x[i];
      sq += x[i] * x[i];
    }
    sm = wred_sum(sm);
    sq = wred_sum(sq);
    const float mu  = sm * (1.0f / D_);
    const float var = sq * (1.0f / D_) - mu * mu;
    const float rs  = rsqrtf(var + 1e-5f);
#pragma unroll
    for (int i = 0; i < 8; ++i) {
      const int c = lane * 8 + i;
      qin_h[(size_t)s * D_ + c] = (half_t)((x[i] - mu) * rs * ln_m_g[c] + ln_m_b[c]);
    }
  }
  __syncthreads();

  // ---- I: hid = relu(LN(slots)@W1^T + b1) -> hidh ----
  for (int nt = wv * 2; nt < wv * 2 + 2; ++nt) {
    v8f acc = zero8();
#pragma unroll
    for (int kc = 0; kc < 8; ++kc) {
      v16h a = frag_h(qin_h + kc * 32, D_, lane);
      v16h b = frag_h(W1_h + (size_t)nt * 16 * D_ + kc * 32, D_, lane);
      acc = wmma16(a, b, acc);
    }
    const int col = nt * 16 + (lane & 15);
    const int mb  = (lane < 16) ? 0 : 8;
#pragma unroll
    for (int r = 0; r < 8; ++r) {
      const float hv = (lane < 16) ? fmaxf(acc[r] + mlp_b1[col], 0.0f) : 0.0f;
      hidh[(size_t)(mb + r) * D_ + col] = (half_t)hv;
    }
  }
  __syncthreads();

  // ---- J: slots += hid@W2^T + b2 ----
  for (int nt = wv * 2; nt < wv * 2 + 2; ++nt) {
    v8f acc = zero8();
#pragma unroll
    for (int kc = 0; kc < 8; ++kc) {
      v16h a = frag_h(hidh + kc * 32, D_, lane);
      v16h b = frag_h(W2_h + (size_t)nt * 16 * D_ + kc * 32, D_, lane);
      acc = wmma16(a, b, acc);
    }
    if (lane < 16) {
      const int col = nt * 16 + lane;
#pragma unroll
      for (int r = 0; r < 8; ++r) sl[(size_t)r * D_ + col] += acc[r] + mlp_b2[col];
    }
  }
  __syncthreads();

  // ---- K: write back ----
  for (int i = tid; i < S_ * D_; i += 256) {
    slots_g[(size_t)n * S_ * D_ + i] = sl[i];
    if (write_out) out[(size_t)n * S_ * D_ + i] = sl[i];
  }
}

// ---------------------------------------------------------------------------
// Host launcher
// ---------------------------------------------------------------------------
extern "C" void kernel_launch(void* const* d_in, const int* in_sizes, int n_in,
                              void* d_out, int out_size, void* d_ws,
                              size_t ws_size, hipStream_t stream) {
  (void)in_sizes; (void)n_in; (void)out_size; (void)ws_size;

  const float* inputs   = (const float*)d_in[0];
  const float* noise    = (const float*)d_in[1];
  const float* slots_mu = (const float*)d_in[2];
  const float* slots_ls = (const float*)d_in[3];
  const float* Wq       = (const float*)d_in[4];
  const float* Wk       = (const float*)d_in[5];
  const float* Wv       = (const float*)d_in[6];
  const float* W_ih     = (const float*)d_in[7];
  const float* W_hh     = (const float*)d_in[8];
  const float* b_ih     = (const float*)d_in[9];
  const float* b_hh     = (const float*)d_in[10];
  const float* mlp_W1   = (const float*)d_in[11];
  const float* mlp_b1   = (const float*)d_in[12];
  const float* mlp_W2   = (const float*)d_in[13];
  const float* mlp_b2   = (const float*)d_in[14];
  const float* ln_in_g  = (const float*)d_in[15];
  const float* ln_in_b  = (const float*)d_in[16];
  const float* ln_s_g   = (const float*)d_in[17];
  const float* ln_s_b   = (const float*)d_in[18];
  const float* ln_m_g   = (const float*)d_in[19];
  const float* ln_m_b   = (const float*)d_in[20];
  float* out = (float*)d_out;

  // ---- workspace layout (bytes) ----
  char* ws = (char*)d_ws;
  const size_t DD  = (size_t)D_ * D_;       // 65536
  const size_t DD3 = (size_t)3 * D_ * D_;   // 196608
  half_t* Wq_h  = (half_t*)(ws + 0);
  half_t* Wk_h  = (half_t*)(ws + 131072);
  half_t* Wv_h  = (half_t*)(ws + 262144);
  half_t* Wih_h = (half_t*)(ws + 393216);
  half_t* Whh_h = (half_t*)(ws + 786432);
  half_t* W1_h  = (half_t*)(ws + 1179648);
  half_t* W2_h  = (half_t*)(ws + 1310720);
  float*  slots = (float*)(ws + 1441792);                  // 64*8*256 f32
  half_t* k_h   = (half_t*)(ws + 1966080);                 // [n][t][d] f16
  half_t* v_t   = (half_t*)(ws + 1966080 + (size_t)134217728);  // [n][d][t] f16

  // 1) weight conversion fp32 -> f16
  sa_cvt_f16<<<(int)((DD + 255) / 256), 256, 0, stream>>>(Wq, Wq_h, (int)DD);
  sa_cvt_f16<<<(int)((DD + 255) / 256), 256, 0, stream>>>(Wk, Wk_h, (int)DD);
  sa_cvt_f16<<<(int)((DD + 255) / 256), 256, 0, stream>>>(Wv, Wv_h, (int)DD);
  sa_cvt_f16<<<(int)((DD3 + 255) / 256), 256, 0, stream>>>(W_ih, Wih_h, (int)DD3);
  sa_cvt_f16<<<(int)((DD3 + 255) / 256), 256, 0, stream>>>(W_hh, Whh_h, (int)DD3);
  sa_cvt_f16<<<(int)((DD + 255) / 256), 256, 0, stream>>>(mlp_W1, W1_h, (int)DD);
  sa_cvt_f16<<<(int)((DD + 255) / 256), 256, 0, stream>>>(mlp_W2, W2_h, (int)DD);

  // 2) slot init
  sa_init_slots<<<(B_ * S_ * D_ + 255) / 256, 256, 0, stream>>>(
      noise, slots_mu, slots_ls, slots);

  // 3) LN + k/v projection (WMMA), v stored transposed
  {
    dim3 grid(T_ / 128, B_);
    sa_ln_kv<<<grid, 256, SMEM_KV, stream>>>(inputs, ln_in_g, ln_in_b, Wk_h,
                                             Wv_h, k_h, v_t);
  }

  // 4) three slot-attention iterations
  for (int it = 0; it < 3; ++it) {
    sa_iter<<<B_, 256, SMEM_ITER, stream>>>(
        slots, k_h, v_t, Wq_h, Wih_h, Whh_h, W1_h, W2_h, b_ih, b_hh, mlp_b1,
        mlp_b2, ln_s_g, ln_s_b, ln_m_g, ln_m_b, out, (it == 2) ? 1 : 0);
  }
}
